// XFormerAttention_37701222924767
// MI455X (gfx1250) — matrix operand order repaired and checked by reading, not compiled, not run on hardware
//
#include <hip/hip_runtime.h>
#include <math.h>

// ProbSparse (Informer-style) attention for MI455X / gfx1250, wave32 + WMMA.
// B=2, T=2048, DM=512, H=8, D=64, u = int(0.4*ln(2048)) = 3.
//
// Precision strategy (MI455X roofline): total FLOPs ~19 GF, memory ~60 MB
// (L2-resident). f32 WMMA (16x16x4) is the slowest matrix path, so the
// largest FLOP block -- the KL *ranking* pass (8.6 GF) -- runs on
// v_wmma_f32_16x16x32_bf16 (8x K per instruction). All math that reaches
// d_out (projections, reduced attention, output GEMM) stays f32.
#define B_  2
#define T_  2048
#define DM_ 512
#define H_  8
#define D_  64
#define U_  3

typedef __attribute__((ext_vector_type(2)))  float  v2f;
typedef __attribute__((ext_vector_type(8)))  float  v8f;
typedef __attribute__((ext_vector_type(8)))  __bf16 v8bf;
typedef __attribute__((ext_vector_type(16))) __bf16 v16bf;

// D = A(16x4,f32) * B(4x16,f32) + C(16x16,f32), one matrix per wave32.
__device__ __forceinline__ v8f wmma_f32_16x16x4(v2f a, v2f b, v8f c) {
  return __builtin_amdgcn_wmma_f32_16x16x4_f32(
      /*neg_a=*/false, a, /*neg_b=*/false, b,
      /*c_mod=*/(short)0, c, /*reuse_a=*/false, /*reuse_b=*/false);
}

__device__ __forceinline__ v8f wmma_f32_16x16x32_bf16(v16bf a, v16bf b, v8f c) {
  return __builtin_amdgcn_wmma_f32_16x16x32_bf16(
      /*neg_a=*/false, a, /*neg_b=*/false, b,
      /*c_mod=*/(short)0, c, /*reuse_a=*/false, /*reuse_b=*/false);
}

// ---------------------------------------------------------------------------
// C[n][m] = sum_k A[n][k] * W[m][k] + bias[m]   (i.e. C = A @ W^T + b)
// A: [N, 512], W: [512, 512], C: [N, 512]. One 16x16 C tile per wave.
// f32 A fragment layout (ISA 7.12.2): lanes 0-15 row M=l hold K=0,1 in the
// two VGPRs, lanes 16-31 hold K=2,3. B mirror-symmetric (N across lanes).
// ---------------------------------------------------------------------------
__global__ void __launch_bounds__(256)
gemm_nt_bias(const float* __restrict__ A, const float* __restrict__ W,
             const float* __restrict__ bias, float* __restrict__ C, int N) {
  const int lane = threadIdx.x & 31;
  const int wave = threadIdx.x >> 5;
  const int tiles_n = DM_ / 16;                       // 32 column tiles
  const int total = (N / 16) * tiles_n;
  const int tile = blockIdx.x * (blockDim.x >> 5) + wave;
  if (tile >= total) return;                          // wave-uniform branch
  const int row0 = (tile / tiles_n) * 16;
  const int col0 = (tile % tiles_n) * 16;
  const int half = lane >> 4;                         // 0: K=0,1  1: K=2,3
  const int l = lane & 15;

  const float* arow = A + (size_t)(row0 + l) * DM_ + half * 2;
  const float* brow = W + (size_t)(col0 + l) * DM_ + half * 2;  // W^T as B

  v8f acc = {};
#pragma unroll 4
  for (int k0 = 0; k0 < DM_; k0 += 4) {
    v2f a = *(const v2f*)(arow + k0);
    v2f b = *(const v2f*)(brow + k0);
    acc = wmma_f32_16x16x4(a, b, acc);
  }
  const float bval = bias[col0 + l];
#pragma unroll
  for (int i = 0; i < 8; ++i) {                       // C/D layout: VGPR i ->
    const int r = row0 + i + half * 8;                // row i (+8 for hi half)
    C[(size_t)r * DM_ + col0 + l] = acc[i] + bval;
  }
}

// ---------------------------------------------------------------------------
// f32 -> bf16 (round-to-nearest-even), with optional exact pow2 pre-scale.
// ---------------------------------------------------------------------------
__global__ void __launch_bounds__(256)
to_bf16_scaled(const float* __restrict__ src, unsigned short* __restrict__ dst,
               float scale, int n) {
  const int i = blockIdx.x * blockDim.x + threadIdx.x;
  if (i < n) {
    union { float f; unsigned u; } v;
    v.f = src[i] * scale;
    const unsigned r = v.u + 0x7FFFu + ((v.u >> 16) & 1u);   // RNE
    dst[i] = (unsigned short)(r >> 16);
  }
}

// ---------------------------------------------------------------------------
// Streaming KL-divergence stats in bf16 WMMA (selection/ranking only).
// For each (b,h) and query row: s_k = q.k/sqrt(D) (1/8 folded into q_bf16,
// exact);  kl = E_p[s] - logsumexp(s) - log(1/T + 1e-10), maintained
// flash-style as (m, Z=sum e^{s-m}, S1=sum s*e^{s-m}) per row; the 2048x2048
// score matrix is never materialized.
//
// bf16 A layout (ISA 7.12.2, 16-bit A 16x32): lane (l,half) holds row M=l,
// K run [half*16, +16) -- contiguous order, not the ISA's; matches B.
// bf16 B layout (32x16): lane (l,half) holds col N=l, K run [half*16, +16)
// -- a single contiguous 32B load since k rows are channel-contiguous.
// ---------------------------------------------------------------------------
__global__ void __launch_bounds__(256)
kl_stats_bf16(const unsigned short* __restrict__ qbf,   // bf16(q/8)
              const unsigned short* __restrict__ kbf,   // bf16(k)
              float* __restrict__ kl) {
  const int lane = threadIdx.x & 31;
  const int wave = threadIdx.x >> 5;
  const int qtiles = T_ / 16;                          // 128
  const int gw = blockIdx.x * (blockDim.x >> 5) + wave;
  if (gw >= B_ * H_ * qtiles) return;
  const int bh = gw / qtiles;
  const int qt = gw % qtiles;
  const int b = bh / H_;
  const int h = bh % H_;
  const int half = lane >> 4;
  const int l = lane & 15;
  const size_t base = (size_t)b * T_ * DM_ + (size_t)h * D_;

  // A fragments (q rows) for the two K-blocks c=[0,32) and c=[32,64):
  // loaded once, held in registers for the whole key stream.
  const __bf16* qrow = (const __bf16*)qbf + base + (size_t)(qt * 16 + l) * DM_;
  const v16bf a0 = *(const v16bf*)(qrow + 0  + half * 16);
  const v16bf a1 = *(const v16bf*)(qrow + 32 + half * 16);

  float m[8], Z[8], S1[8];
#pragma unroll
  for (int i = 0; i < 8; ++i) { m[i] = -INFINITY; Z[i] = 0.f; S1[i] = 0.f; }

  const __bf16* kbase = (const __bf16*)kbf + base + half * 16;
  for (int kt = 0; kt < T_; kt += 16) {
    const __bf16* krow = kbase + (size_t)(kt + l) * DM_;
    if (kt + 16 < T_)
      __builtin_prefetch(krow + 16 * DM_, 0, 1);       // global_prefetch_b8
    const v16bf b0 = *(const v16bf*)(krow);            // K=[half*16, +16)
    const v16bf b1 = *(const v16bf*)(krow + 32);       // K=[32+half*16, +16)
    v8f acc = {};
    acc = wmma_f32_16x16x32_bf16(a0, b0, acc);
    acc = wmma_f32_16x16x32_bf16(a1, b1, acc);
    // Online per-row stats over this lane's 16-key column slice.
#pragma unroll
    for (int i = 0; i < 8; ++i) {
      const float s = acc[i];
      const float mn = fmaxf(m[i], s);
      const float eo = __expf(m[i] - mn);              // exp(-inf)=0 first time
      const float en = __expf(s - mn);
      Z[i]  = Z[i]  * eo + en;
      S1[i] = S1[i] * eo + s * en;
      m[i]  = mn;
    }
  }
  // Merge the 16 lanes that share each matrix row (within each wave half).
#pragma unroll
  for (int off = 1; off < 16; off <<= 1) {
#pragma unroll
    for (int i = 0; i < 8; ++i) {
      const float mo = __shfl_xor(m[i],  off, 32);
      const float Zo = __shfl_xor(Z[i],  off, 32);
      const float So = __shfl_xor(S1[i], off, 32);
      const float mn = fmaxf(m[i], mo);
      const float ea = __expf(m[i] - mn);
      const float eb = __expf(mo   - mn);
      Z[i]  = Z[i]  * ea + Zo * eb;
      S1[i] = S1[i] * ea + So * eb;
      m[i]  = mn;
    }
  }
  if (l == 0) {
    const float log_uniform = logf(1.0f / (float)T_ + 1e-10f);
#pragma unroll
    for (int i = 0; i < 8; ++i) {
      const int row = qt * 16 + i + half * 8;
      const float lse = m[i] + logf(Z[i]);
      kl[(size_t)bh * T_ + row] = S1[i] / Z[i] - lse - log_uniform;
    }
  }
}

// ---------------------------------------------------------------------------
// Top-3 KL queries per (b,h); jax.lax.top_k tie-break = lower index first.
// ---------------------------------------------------------------------------
__global__ void __launch_bounds__(256)
topk3_kernel(const float* __restrict__ kl, int* __restrict__ topk) {
  __shared__ float svals[256 * U_];
  __shared__ int   sidx [256 * U_];
  const int bh = blockIdx.x;
  const float* vals = kl + (size_t)bh * T_;
  const int t = threadIdx.x;
  float bv[U_] = {-INFINITY, -INFINITY, -INFINITY};
  int   bi[U_] = {T_, T_, T_};
  for (int i = t; i < T_; i += 256) {                  // strict >: first index kept
    const float x = vals[i];
    if (x > bv[0])      { bv[2]=bv[1]; bi[2]=bi[1]; bv[1]=bv[0]; bi[1]=bi[0]; bv[0]=x; bi[0]=i; }
    else if (x > bv[1]) { bv[2]=bv[1]; bi[2]=bi[1]; bv[1]=x; bi[1]=i; }
    else if (x > bv[2]) { bv[2]=x; bi[2]=i; }
  }
  for (int j = 0; j < U_; ++j) { svals[t*U_+j] = bv[j]; sidx[t*U_+j] = bi[j]; }
  __syncthreads();
  if (t == 0) {
    float fv[U_] = {-INFINITY, -INFINITY, -INFINITY};
    int   fi[U_] = {T_, T_, T_};
    for (int n = 0; n < 256 * U_; ++n) {
      const float x = svals[n]; const int ix = sidx[n];
      if (x > fv[0] || (x == fv[0] && ix < fi[0])) {
        fv[2]=fv[1]; fi[2]=fi[1]; fv[1]=fv[0]; fi[1]=fi[0]; fv[0]=x; fi[0]=ix;
      } else if (x > fv[1] || (x == fv[1] && ix < fi[1])) {
        fv[2]=fv[1]; fi[2]=fi[1]; fv[1]=x; fi[1]=ix;
      } else if (x > fv[2] || (x == fv[2] && ix < fi[2])) {
        fv[2]=x; fi[2]=ix;
      }
    }
    for (int j = 0; j < U_; ++j) topk[bh*U_+j] = fi[j];
  }
}

__global__ void zero_f32(float* __restrict__ p, int n) {
  const int i = blockIdx.x * blockDim.x + threadIdx.x;
  if (i < n) p[i] = 0.f;
}

// ---------------------------------------------------------------------------
// Exact f32 reduced attention for the 3 selected queries of one (b,h);
// scatter into zeroed ctx at the selected row/head slot. One block per
// (b,h,u). Matches reference: nan_to_num(-1e4) + clip(+-1e4) + softmax.
// ---------------------------------------------------------------------------
__global__ void __launch_bounds__(256)
sparse_attn(const float* __restrict__ q, const float* __restrict__ k,
            const float* __restrict__ v, const int* __restrict__ topk,
            float* __restrict__ ctx) {
  __shared__ float s_s[T_];                            // 8 KB score row
  __shared__ float qs[D_];
  __shared__ float part[4][D_];
  __shared__ float rbuf[8];
  __shared__ float bcast;

  const int bh = blockIdx.x / U_;
  const int ui = blockIdx.x % U_;
  const int b = bh / H_;
  const int h = bh % H_;
  const int qidx = topk[bh * U_ + ui];
  const size_t base = (size_t)b * T_ * DM_ + (size_t)h * D_;
  const int t = threadIdx.x;

  if (t < D_) qs[t] = q[base + (size_t)qidx * DM_ + t];
  __syncthreads();

  float lmax = -INFINITY;
  for (int key = t; key < T_; key += 256) {
    const float* kr = k + base + (size_t)key * DM_;
    float dot = 0.f;
#pragma unroll
    for (int c = 0; c < D_; ++c) dot += qs[c] * kr[c];
    float s = dot * 0.125f;
    if (isnan(s)) s = -10000.0f;                       // nan_to_num + clip
    s = fminf(fmaxf(s, -10000.0f), 10000.0f);
    s_s[key] = s;
    lmax = fmaxf(lmax, s);
  }
  for (int off = 16; off > 0; off >>= 1)
    lmax = fmaxf(lmax, __shfl_xor(lmax, off, 32));
  if ((t & 31) == 0) rbuf[t >> 5] = lmax;
  __syncthreads();
  if (t == 0) {
    float mm = rbuf[0];
    for (int i = 1; i < 8; ++i) mm = fmaxf(mm, rbuf[i]);
    bcast = mm;
  }
  __syncthreads();
  const float m = bcast;

  float lsum = 0.f;
  for (int key = t; key < T_; key += 256) lsum += __expf(s_s[key] - m);
  for (int off = 16; off > 0; off >>= 1) lsum += __shfl_xor(lsum, off, 32);
  __syncthreads();
  if ((t & 31) == 0) rbuf[t >> 5] = lsum;
  __syncthreads();
  if (t == 0) {
    float z = 0.f;
    for (int i = 0; i < 8; ++i) z += rbuf[i];
    bcast = z;
  }
  __syncthreads();
  const float Z = bcast;

  const int d = t & (D_ - 1);                          // output dim
  const int g = t >> 6;                                // key group 0..3
  float acc = 0.f;
  for (int key = g; key < T_; key += 4)
    acc += __expf(s_s[key] - m) * v[base + (size_t)key * DM_ + d];
  part[g][d] = acc;
  __syncthreads();
  if (t < D_) {
    const float o = (part[0][t] + part[1][t] + part[2][t] + part[3][t]) / Z;
    ctx[base + (size_t)qidx * DM_ + t] = o;
  }
}

// ---------------------------------------------------------------------------
extern "C" void kernel_launch(void* const* d_in, const int* in_sizes, int n_in,
                              void* d_out, int out_size, void* d_ws, size_t ws_size,
                              hipStream_t stream) {
  (void)in_sizes; (void)n_in; (void)out_size; (void)ws_size;
  const float* query = (const float*)d_in[0];
  const float* keyx  = (const float*)d_in[1];
  const float* value = (const float*)d_in[2];
  const float* Wq    = (const float*)d_in[3];
  const float* bq    = (const float*)d_in[4];
  const float* Wk    = (const float*)d_in[5];
  const float* bk    = (const float*)d_in[6];
  const float* Wv    = (const float*)d_in[7];
  const float* bv    = (const float*)d_in[8];
  const float* Wo    = (const float*)d_in[9];
  const float* bo    = (const float*)d_in[10];

  const size_t NTOK = (size_t)B_ * T_;                 // 4096 rows
  const size_t NELT = NTOK * DM_;                      // 2M elements
  float* qb   = (float*)d_ws;
  float* kb   = qb + NELT;
  float* vb   = kb + NELT;
  float* klb  = vb + NELT;                             // B*H*T floats
  int*   tkb  = (int*)(klb + (size_t)B_ * H_ * T_);
  float* ctx  = (float*)((char*)tkb + 256);            // padded/aligned
  unsigned short* qbf = (unsigned short*)(ctx + NELT); // bf16(q/8)
  unsigned short* kbf = qbf + NELT;                    // bf16(k)

  const dim3 blk(256);
  const int gemm_blocks = (int)((NTOK / 16) * (DM_ / 16) / 8);   // 1024
  const int cvt_blocks  = (int)(NELT / 256);                     // 8192

  // Q/K/V projections (C = X @ W^T + b), exact f32 WMMA.
  gemm_nt_bias<<<gemm_blocks, blk, 0, stream>>>(query, Wq, bq, qb, (int)NTOK);
  gemm_nt_bias<<<gemm_blocks, blk, 0, stream>>>(keyx,  Wk, bk, kb, (int)NTOK);
  gemm_nt_bias<<<gemm_blocks, blk, 0, stream>>>(value, Wv, bv, vb, (int)NTOK);

  // bf16 copies for the ranking pass (1/sqrt(D)=0.125 folded into q, exact).
  to_bf16_scaled<<<cvt_blocks, blk, 0, stream>>>(qb, qbf, 0.125f, (int)NELT);
  to_bf16_scaled<<<cvt_blocks, blk, 0, stream>>>(kb, kbf, 1.0f,   (int)NELT);

  // Streaming KL stats on bf16 WMMA: B*H*(T/16)=2048 waves, 8 waves/block.
  kl_stats_bf16<<<256, blk, 0, stream>>>(qbf, kbf, klb);

  // Top-3 query selection per (b,h).
  topk3_kernel<<<B_ * H_, blk, 0, stream>>>(klb, tkb);

  // Zero ctx, then exact f32 reduced attention scattered into it.
  zero_f32<<<(int)(NELT / 256), blk, 0, stream>>>(ctx, (int)NELT);
  sparse_attn<<<B_ * H_ * U_, blk, 0, stream>>>(qb, kb, vb, tkb, ctx);

  // Output projection into d_out, exact f32 WMMA.
  gemm_nt_bias<<<gemm_blocks, blk, 0, stream>>>(ctx, Wo, bo, (float*)d_out, (int)NTOK);
}
